// QuantizedLinear_22058952032839
// MI455X (gfx1250) — matrix operand order repaired and checked by reading, not compiled, not run on hardware
//
#include <hip/hip_runtime.h>
#include <hip/hip_fp16.h>
#include <stdint.h>

// ---- problem constants (match reference) ----
#define TOKENS 4096
#define IN_F   4096
#define OUT_F  4096
#define GROUP  128
#define NGRP   (IN_F / GROUP)   // 32

typedef __attribute__((ext_vector_type(16))) _Float16 v16h;
typedef __attribute__((ext_vector_type(8)))  _Float16 v8h;
typedef __attribute__((ext_vector_type(8)))  float    v8f;
typedef __attribute__((ext_vector_type(4)))  float    v4f;
typedef __attribute__((ext_vector_type(4)))  int      v4i;

// ---- CDNA5 async global->LDS support (guarded; fall back to staged path) ----
#if __has_builtin(__builtin_amdgcn_global_load_async_to_lds_b128)
#define HAVE_ASYNC_LDS 1
typedef __attribute__((address_space(1))) v4i gl_v4i;   // global memory
typedef __attribute__((address_space(3))) v4i ds_v4i;   // LDS
template <int OFS>
static __device__ __forceinline__ void async_b128(const _Float16* g, _Float16* l) {
    // INST_OFFSET applies to both the global and LDS addresses (ISA 15.18)
    __builtin_amdgcn_global_load_async_to_lds_b128(
        (gl_v4i*)g, (ds_v4i*)l, /*offset=*/OFS, /*cpol=*/0);
}
#endif

static __device__ __forceinline__ void wait_async0() {
#if __has_builtin(__builtin_amdgcn_s_wait_asynccnt)
    __builtin_amdgcn_s_wait_asynccnt(0);
#else
    asm volatile("s_wait_asynccnt 0x0" ::: "memory");
#endif
}

// =====================================================================
// Pass 1: x fp32 -> fp16   (memory bound, one-shot)
// =====================================================================
__global__ void __launch_bounds__(256)
cvt_x_f16(const float* __restrict__ X, _Float16* __restrict__ Xh)
{
    const size_t i = ((size_t)blockIdx.x * 256 + threadIdx.x) * 16;
    const v4f f0 = ((const v4f*)(X + i))[0];
    const v4f f1 = ((const v4f*)(X + i))[1];
    const v4f f2 = ((const v4f*)(X + i))[2];
    const v4f f3 = ((const v4f*)(X + i))[3];
    v8h h0, h1;
#pragma unroll
    for (int e = 0; e < 4; ++e) {
        h0[e]     = (_Float16)f0[e];
        h0[4 + e] = (_Float16)f1[e];
        h1[e]     = (_Float16)f2[e];
        h1[4 + e] = (_Float16)f3[e];
    }
    *(v8h*)(Xh + i)     = h0;
    *(v8h*)(Xh + i + 8) = h1;
}

// =====================================================================
// Pass 2: W int8 * group scale -> fp16 (same f16 math as the reference)
// =====================================================================
__global__ void __launch_bounds__(256)
dequant_w_f16(const int8_t* __restrict__ W, const _Float16* __restrict__ S,
              _Float16* __restrict__ Wh)
{
    const size_t i = ((size_t)blockIdx.x * 256 + threadIdx.x) * 16;
    const int row = (int)(i >> 12);          // / IN_F
    const int k   = (int)(i & (IN_F - 1));   // 16-elem chunk stays inside one group
    const _Float16 s = S[row * NGRP + (k >> 7)];
    union { v4i v; int8_t b[16]; } w;
    w.v = *(const v4i*)(W + i);
    v8h h0, h1;
#pragma unroll
    for (int e = 0; e < 8; ++e) {
        h0[e] = (_Float16)w.b[e]     * s;
        h1[e] = (_Float16)w.b[8 + e] * s;
    }
    *(v8h*)(Wh + i)     = h0;
    *(v8h*)(Wh + i + 8) = h1;
}

// =====================================================================
// Main GEMM: fp16 x fp16 -> fp32, WMMA 16x16x32
//   block = 128(M) x 256(N), 8 wave32s in a 2x4 grid, 64x64 per wave
// =====================================================================
#define BM  128
#define BN  256
#define BK  32
#define PAD 8
#define LDK (BK + PAD)          // 40 halves = 80B row stride (16B aligned, 20-bank step)
#define NK  (IN_F / BK)         // 128 K-steps

__global__ void __launch_bounds__(256)
gemm_wmma_f16(const _Float16* __restrict__ Xh,   // [TOKENS, IN_F] f16
              const _Float16* __restrict__ Wh,   // [OUT_F, IN_F] f16
              const _Float16* __restrict__ Bv,   // [OUT_F] f16
              float* __restrict__ O)             // [TOKENS, OUT_F] f32
{
    __shared__ __align__(16) _Float16 As[2][BM][LDK];  // 20 KB
    __shared__ __align__(16) _Float16 Bs[2][BN][LDK];  // 40 KB

    const int tid  = threadIdx.x;
    const int lane = tid & 31;
    const int wave = tid >> 5;
    const int wm   = wave >> 2;          // 0..1 : 64-row slab
    const int wn   = wave & 3;           // 0..3 : 64-col slab
    const int m0   = blockIdx.y * BM;
    const int n0   = blockIdx.x * BN;

    // ---- cooperative tile-load mapping ----
    const int tr  = tid >> 1;            // 0..127 (A row)
    const int tca = (tid & 1) * 16;      // A half-row chunk
    const _Float16* pA = Xh + (size_t)(m0 + tr) * IN_F + tca;
    const _Float16* pB = Wh + (size_t)(n0 + tid) * IN_F;   // B row = tid (0..255)

    // ---- accumulators: 4 x 4 tiles of 16x16 f32 ----
    v8f acc[4][4];
#pragma unroll
    for (int i = 0; i < 4; ++i)
#pragma unroll
        for (int j = 0; j < 4; ++j)
            acc[i][j] = (v8f)0.0f;

#if defined(HAVE_ASYNC_LDS)
    // -------- async global->LDS tile mover (no VGPR staging) --------
    auto load_tile = [&](int kt, int buf) {
        const _Float16* xa = pA + (size_t)kt * BK;
        _Float16* ap = &As[buf][tr][tca];
        async_b128<0> (xa, ap);
        async_b128<16>(xa, ap);
        const _Float16* xb = pB + (size_t)kt * BK;
        _Float16* bp = &Bs[buf][tid][0];
        async_b128<0> (xb, bp);
        async_b128<16>(xb, bp);
        async_b128<32>(xb, bp);
        async_b128<48>(xb, bp);
    };
#else
    // -------- fallback: register-staged tile mover --------
    v8h a0, a1, b0, b1, b2, b3;
    auto gload = [&](int kt) {
        const _Float16* xa = pA + (size_t)kt * BK;
        a0 = ((const v8h*)xa)[0];
        a1 = ((const v8h*)xa)[1];
        const _Float16* xb = pB + (size_t)kt * BK;
        b0 = ((const v8h*)xb)[0];
        b1 = ((const v8h*)xb)[1];
        b2 = ((const v8h*)xb)[2];
        b3 = ((const v8h*)xb)[3];
    };
    auto lstore = [&](int buf) {
        _Float16* ap = &As[buf][tr][tca];
        *(v8h*)ap       = a0;
        *(v8h*)(ap + 8) = a1;
        _Float16* bp = &Bs[buf][tid][0];
        *(v8h*)bp        = b0;
        *(v8h*)(bp + 8)  = b1;
        *(v8h*)(bp + 16) = b2;
        *(v8h*)(bp + 24) = b3;
    };
#endif

    const int lrow = lane & 15;
    const int lhi  = lane >> 4;

    // compute: preload 4 B fragments, then stream A fragments one at a time.
    // Keeps live set ~ acc(128) + bf(32) + a(8) + addressing < 256 VGPRs so the
    // allocator never crosses into MSB VGPR space (no v_mov_b64 shuffling).
    auto compute = [&](int buf) {
        const int ka = lhi * 8;
        const int kb = lhi * 8;
        v16h bf[4];
#pragma unroll
        for (int j = 0; j < 4; ++j) {
            const _Float16* p = &Bs[buf][wn * 64 + j * 16 + lrow][kb];
            v8h lo = *(const v8h*)p;
            v8h hi = *(const v8h*)(p + 16);
            bf[j] = __builtin_shufflevector(lo, hi, 0, 1, 2, 3, 4, 5, 6, 7,
                                            8, 9, 10, 11, 12, 13, 14, 15);
        }
#pragma unroll
        for (int i = 0; i < 4; ++i) {
            const _Float16* p = &As[buf][wm * 64 + i * 16 + lrow][ka];
            v8h lo = *(const v8h*)p;
            v8h hi = *(const v8h*)(p + 16);
            v16h a = __builtin_shufflevector(lo, hi, 0, 1, 2, 3, 4, 5, 6, 7,
                                             8, 9, 10, 11, 12, 13, 14, 15);
#pragma unroll
            for (int j = 0; j < 4; ++j)
                acc[i][j] = __builtin_amdgcn_wmma_f32_16x16x32_f16(
                    false, a, false, bf[j], (short)0, acc[i][j], false, false);
        }
    };

    // ---- double-buffered K loop, ONE barrier per iteration, no unroll ----
#if defined(HAVE_ASYNC_LDS)
    load_tile(0, 0);
    wait_async0();
    __syncthreads();
#pragma unroll 1
    for (int kt = 0; kt < NK - 1; ++kt) {
        load_tile(kt + 1, (kt + 1) & 1);   // overlaps compute
        compute(kt & 1);
        wait_async0();                     // my async writes are done
        __syncthreads();                   // publish next buffer
    }
    compute((NK - 1) & 1);
#else
    gload(0);
    lstore(0);
    __syncthreads();
#pragma unroll 1
    for (int kt = 0; kt < NK - 1; ++kt) {
        gload(kt + 1);
        compute(kt & 1);
        lstore((kt + 1) & 1);              // buffer free since barrier at kt-1
        __syncthreads();                   // publish for iteration kt+1
    }
    compute((NK - 1) & 1);
#endif

    // ---- epilogue: bias + fp32 store ----
    const int mbase = m0 + wm * 64 + lhi * 8;
    const int nbase = n0 + wn * 64 + lrow;
#pragma unroll
    for (int i = 0; i < 4; ++i) {
#pragma unroll
        for (int j = 0; j < 4; ++j) {
            const int col = nbase + j * 16;
            const float bv = (float)Bv[col];
            const int row = mbase + i * 16;
#pragma unroll
            for (int r = 0; r < 8; ++r)
                O[(size_t)(row + r) * OUT_F + col] = acc[i][j][r] + bv;
        }
    }
}

// =====================================================================
// Fallback: fused dequant GEMM (if d_ws is too small)
// =====================================================================
#define FBM 128
#define FBN 128
#define FLDK (BK + PAD)

__global__ void __launch_bounds__(256)
qlinear_wmma_fused(const float* __restrict__ X, const int8_t* __restrict__ W,
                   const _Float16* __restrict__ S, const _Float16* __restrict__ Bv,
                   float* __restrict__ O)
{
    __shared__ __align__(16) _Float16 As[2][FBM][FLDK];
    __shared__ __align__(16) _Float16 Bs[2][FBN][FLDK];

    const int tid  = threadIdx.x;
    const int lane = tid & 31;
    const int wave = tid >> 5;
    const int wm   = wave >> 1;
    const int wn   = wave & 1;
    const int m0   = blockIdx.y * FBM;
    const int n0   = blockIdx.x * FBN;
    const int tr = tid >> 1;
    const int tc = tid & 1;

    const float*    xg = X + (size_t)(m0 + tr) * IN_F + tc * 16;
    const int8_t*   wg = W + (size_t)(n0 + tr) * IN_F + tc * 16;
    const _Float16* sg = S + (size_t)(n0 + tr) * NGRP;

    v8f acc[2][4];
#pragma unroll
    for (int i = 0; i < 2; ++i)
#pragma unroll
        for (int j = 0; j < 4; ++j) acc[i][j] = (v8f)0.0f;

    v4f xs0, xs1, xs2, xs3;
    union { v4i v; int8_t b[16]; } wsu;
    float sf = 0.0f;

    auto gload = [&](int kt) {
        const float* xp = xg + (size_t)kt * BK;
        xs0 = ((const v4f*)xp)[0]; xs1 = ((const v4f*)xp)[1];
        xs2 = ((const v4f*)xp)[2]; xs3 = ((const v4f*)xp)[3];
        wsu.v = *((const v4i*)(wg + (size_t)kt * BK));
        sf = (float)sg[(kt * BK) / GROUP];
    };
    auto lstore = [&](int buf) {
        v8h a0, a1;
#pragma unroll
        for (int e = 0; e < 4; ++e) {
            a0[e] = (_Float16)xs0[e]; a0[4 + e] = (_Float16)xs1[e];
            a1[e] = (_Float16)xs2[e]; a1[4 + e] = (_Float16)xs3[e];
        }
        _Float16* ap = &As[buf][tr][tc * 16];
        *(v8h*)ap = a0; *(v8h*)(ap + 8) = a1;
        v8h c0, c1;
#pragma unroll
        for (int e = 0; e < 8; ++e) {
            c0[e] = (_Float16)(sf * (float)wsu.b[e]);
            c1[e] = (_Float16)(sf * (float)wsu.b[8 + e]);
        }
        _Float16* bp = &Bs[buf][tr][tc * 16];
        *(v8h*)bp = c0; *(v8h*)(bp + 8) = c1;
    };

    const int lrow = lane & 15;
    const int lhi  = lane >> 4;
    auto compute = [&](int buf) {
        const int ka = lhi * 8;
        const int kb = lhi * 8;
        v16h bf[4];
#pragma unroll
        for (int j = 0; j < 4; ++j) {
            const _Float16* p = &Bs[buf][wn * 64 + j * 16 + lrow][kb];
            v8h lo = *(const v8h*)p;
            v8h hi = *(const v8h*)(p + 16);
            bf[j] = __builtin_shufflevector(lo, hi, 0, 1, 2, 3, 4, 5, 6, 7,
                                            8, 9, 10, 11, 12, 13, 14, 15);
        }
#pragma unroll
        for (int i = 0; i < 2; ++i) {
            const _Float16* p = &As[buf][wm * 32 + i * 16 + lrow][ka];
            v8h lo = *(const v8h*)p;
            v8h hi = *(const v8h*)(p + 16);
            v16h a = __builtin_shufflevector(lo, hi, 0, 1, 2, 3, 4, 5, 6, 7,
                                             8, 9, 10, 11, 12, 13, 14, 15);
#pragma unroll
            for (int j = 0; j < 4; ++j)
                acc[i][j] = __builtin_amdgcn_wmma_f32_16x16x32_f16(
                    false, a, false, bf[j], (short)0, acc[i][j], false, false);
        }
    };

    gload(0); lstore(0); __syncthreads();
#pragma unroll 1
    for (int kt = 0; kt < NK - 1; ++kt) {
        gload(kt + 1);
        compute(kt & 1);
        lstore((kt + 1) & 1);
        __syncthreads();
    }
    compute((NK - 1) & 1);

    const int mbase = m0 + wm * 32 + lhi * 8;
    const int nbase = n0 + wn * 64 + lrow;
#pragma unroll
    for (int i = 0; i < 2; ++i)
#pragma unroll
        for (int j = 0; j < 4; ++j) {
            const int col = nbase + j * 16;
            const float bv = (float)Bv[col];
            const int row = mbase + i * 16;
#pragma unroll
            for (int r = 0; r < 8; ++r)
                O[(size_t)(row + r) * OUT_F + col] = acc[i][j][r] + bv;
        }
}

// =====================================================================
// Host entry
// =====================================================================
extern "C" void kernel_launch(void* const* d_in, const int* in_sizes, int n_in,
                              void* d_out, int out_size, void* d_ws, size_t ws_size,
                              hipStream_t stream) {
    (void)in_sizes; (void)n_in; (void)out_size;
    const float*    x  = (const float*)d_in[0];
    const int8_t*   w  = (const int8_t*)d_in[1];
    const _Float16* s  = (const _Float16*)d_in[2];
    const _Float16* bv = (const _Float16*)d_in[3];
    float* out = (float*)d_out;

    const size_t nX = (size_t)TOKENS * IN_F;   // 16M halves = 32 MB
    const size_t nW = (size_t)OUT_F  * IN_F;   // 16M halves = 32 MB

    if (ws_size >= (nX + nW) * sizeof(_Float16)) {
        _Float16* Xh = (_Float16*)d_ws;
        _Float16* Wh = Xh + nX;
        cvt_x_f16    <<<dim3((unsigned)(nX / 16 / 256)), 256, 0, stream>>>(x, Xh);
        dequant_w_f16<<<dim3((unsigned)(nW / 16 / 256)), 256, 0, stream>>>(w, s, Wh);
        dim3 grid(OUT_F / BN, TOKENS / BM);    // 16 x 32
        gemm_wmma_f16<<<grid, 256, 0, stream>>>(Xh, Wh, bv, out);
    } else {
        dim3 grid(OUT_F / FBN, TOKENS / FBM);  // 32 x 32
        qlinear_wmma_fused<<<grid, 256, 0, stream>>>(x, w, s, bv, out);
    }
}
